// ParallelBlock_24498493456704
// MI455X (gfx1250) — matrix operand order repaired and checked by reading, not compiled, not run on hardware
//
#include <hip/hip_runtime.h>
#include <hip/hip_bf16.h>

// ---------------------------------------------------------------------------
// MI455X / gfx1250: wave32, WMMA 16x16x32 bf16 (fp32 accumulate)
// ---------------------------------------------------------------------------

typedef __attribute__((ext_vector_type(16))) __bf16 v16bf;
typedef __attribute__((ext_vector_type(8)))  float  v8f;
typedef __attribute__((ext_vector_type(4)))  int    v4i;
typedef unsigned int  __attribute__((may_alias)) u32a;

union AFrag { v16bf v; unsigned u[8]; };
union CFrag { v8f   v; float    f[8]; };

#define DIM     1024
#define NTOK    4096      // B*N = 4*1024
#define HEADS   16
#define HD      64        // head dim
#define NQKV    3788      // 3*DIM + 716
#define HID     716
#define HIDP    736       // HID padded to multiple of 32
#define D2      2048

#ifndef __has_builtin
#define __has_builtin(x) 0
#endif
#if __has_builtin(__builtin_amdgcn_global_load_async_to_lds_b128)
#define HAVE_ASYNC_LDS 1
#else
#define HAVE_ASYNC_LDS 0
#endif

__device__ __forceinline__ unsigned short f2bf(float f) {
    unsigned u = __builtin_bit_cast(unsigned, f);
    unsigned r = (u + 0x7FFFu + ((u >> 16) & 1u)) >> 16;   // RNE
    return (unsigned short)r;
}

// 16-byte global -> LDS copy: async on gfx1250 (ASYNCcnt) with sync fallback
__device__ __forceinline__ void copy16_g2l(const unsigned short* g, unsigned short* l) {
#if HAVE_ASYNC_LDS
    typedef v4i __attribute__((address_space(1)))* gv4p;   // pointer to v4i in global
    typedef v4i __attribute__((address_space(3)))* lv4p;   // pointer to v4i in LDS
    __builtin_amdgcn_global_load_async_to_lds_b128((gv4p)g, (lv4p)l, 0, 0);
#else
    typedef uint4 __attribute__((may_alias)) uint4a;
    *(uint4a*)l = *(const uint4a*)g;
#endif
}

__device__ __forceinline__ void async_drain() {
#if HAVE_ASYNC_LDS
#if __has_builtin(__builtin_amdgcn_s_wait_asynccnt)
    __builtin_amdgcn_s_wait_asynccnt(0);
#else
    asm volatile("s_wait_asynccnt 0x0" ::: "memory");
#endif
#endif
}

// ---------------------------------------------------------------------------
// Weight convert + transpose: Bt[n*Kp + k] = (k<K ? W[k*N+n] : 0) as bf16
// ---------------------------------------------------------------------------
__global__ __launch_bounds__(256)
void cvt_transpose_kernel(const float* __restrict__ W, unsigned short* __restrict__ Bt,
                          int K, int N, int Kp) {
    size_t i = (size_t)blockIdx.x * 256 + threadIdx.x;
    if (i >= (size_t)N * Kp) return;
    int n = (int)(i / Kp);
    int k = (int)(i % Kp);
    float v = (k < K) ? W[(size_t)k * N + n] : 0.f;
    Bt[i] = f2bf(v);
}

// ---------------------------------------------------------------------------
// LayerNorm over DIM, one wave per row, output bf16
// ---------------------------------------------------------------------------
__global__ __launch_bounds__(256)
void ln_kernel(const float* __restrict__ x, const float* __restrict__ g,
               const float* __restrict__ b, unsigned short* __restrict__ outb) {
    const int wave = threadIdx.x >> 5;
    const int lane = threadIdx.x & 31;
    const int row  = blockIdx.x * 8 + wave;
    const float* xr = x + (size_t)row * DIM;

    float s = 0.f;
    #pragma unroll
    for (int i = 0; i < DIM / 32; ++i) s += xr[lane + 32 * i];
    #pragma unroll
    for (int off = 1; off < 32; off <<= 1) s += __shfl_xor(s, off, 32);
    const float mean = s * (1.f / DIM);

    float vs = 0.f;
    #pragma unroll
    for (int i = 0; i < DIM / 32; ++i) {
        float d = xr[lane + 32 * i] - mean;
        vs += d * d;
    }
    #pragma unroll
    for (int off = 1; off < 32; off <<= 1) vs += __shfl_xor(vs, off, 32);
    const float inv = rsqrtf(vs * (1.f / DIM) + 1e-5f);

    unsigned short* orow = outb + (size_t)row * DIM;
    #pragma unroll
    for (int i = 0; i < DIM / 32; ++i) {
        int e = lane + 32 * i;
        orow[e] = f2bf((xr[e] - mean) * inv * g[e] + b[e]);
    }
}

// ---------------------------------------------------------------------------
// bf16 WMMA GEMM with LDS-staged (double buffered, async) B operand.
//   C[M x N] = A[M x K] @ B[K x N], B pre-transposed as Bt[N][Kp].
// Block = 256 thr (8 waves: 4M x 2N), block tile 128M x 128N,
// wave tile 32M x 64N (2x4 WMMA accumulators, 8 wmma per k-step).
// EPI 0: fp32 store + bias (f1)
// EPI 1: bf16 store + bias into concat buffer (ldc/col_off)
// EPI 2: out = resid + (acc + bias) * ls   (final, fp32)
// ---------------------------------------------------------------------------
#define BROW 40              // LDS row stride in shorts (80B: 16B-aligned, padded)

template<int EPI>
__global__ __launch_bounds__(256)
void gemm_bf16_kernel(const unsigned short* __restrict__ A, int lda,
                      const unsigned short* __restrict__ Bt, int ldb,
                      const float* __restrict__ bias,
                      int N, int K,
                      float* __restrict__ Cf, unsigned short* __restrict__ Cb,
                      int ldc, int col_off,
                      const float* __restrict__ resid, const float* __restrict__ ls) {
    __shared__ unsigned short sB[2][128 * BROW];   // 2 x 128 rows x 32 k (padded)

    const int lane = threadIdx.x & 31;
    const int wave = threadIdx.x >> 5;
    const int hf   = lane >> 4;          // half of wave (0/1)
    const int ln   = lane & 15;
    const int mBase = blockIdx.y * 128 + (wave & 3) * 32;
    const int nBlk  = blockIdx.x * 128;
    const int nWave = (wave >> 2) * 64;

    CFrag c[2][4];
    #pragma unroll
    for (int mi = 0; mi < 2; ++mi)
        #pragma unroll
        for (int ni = 0; ni < 4; ++ni)
            #pragma unroll
            for (int r = 0; r < 8; ++r) c[mi][ni].f[r] = 0.f;

    // cooperative B-tile fill: 512 x 16B chunks, 2 per thread
    auto fill = [&](int buf, int kk) {
        #pragma unroll
        for (int cch = 0; cch < 2; ++cch) {
            int chunk = threadIdx.x + 256 * cch;       // 0..511
            int n = chunk >> 2;                        // 0..127
            int q = chunk & 3;                         // 16B quarter of the 64B row
            int gn = nBlk + n;
            if (gn >= N) gn = N - 1;                   // clamp; results never stored
            copy16_g2l(Bt + (size_t)gn * ldb + kk + q * 8, &sB[buf][n * BROW + q * 8]);
        }
    };

    const int S = K / 32;
    fill(0, 0);
    async_drain();
    __syncthreads();

    for (int s = 0; s < S; ++s) {
        const int kk = s * 32;
        if (s + 1 < S) fill((s + 1) & 1, kk + 32);

        // A fragments (global; L2/WGP$-resident activations)
        AFrag a[2];
        #pragma unroll
        for (int mi = 0; mi < 2; ++mi) {
            const unsigned short* ap = A + (size_t)(mBase + mi * 16 + ln) * lda + kk;
            __builtin_prefetch(ap + 32, 0, 1);
            #pragma unroll
            for (int v = 0; v < 8; ++v) {
                int k = 16 * hf + 2 * v;
                a[mi].u[v] = *(const u32a*)(ap + k);
            }
        }
        // B fragments from LDS, 8 wmma per k-step
        #pragma unroll
        for (int ni = 0; ni < 4; ++ni) {
            AFrag b;
            const u32a* bp = (const u32a*)&sB[s & 1][(nWave + ni * 16 + ln) * BROW + 16 * hf];
            #pragma unroll
            for (int v = 0; v < 8; ++v) b.u[v] = bp[v];
            #pragma unroll
            for (int mi = 0; mi < 2; ++mi)
                c[mi][ni].v = __builtin_amdgcn_wmma_f32_16x16x32_bf16(
                    false, a[mi].v, false, b.v, (short)0, c[mi][ni].v, false, false);
        }

        async_drain();        // own stage-(s+1) async copies complete
        __syncthreads();      // all waves done with buffer s before overwrite
    }

    #pragma unroll
    for (int mi = 0; mi < 2; ++mi)
        #pragma unroll
        for (int ni = 0; ni < 4; ++ni)
            #pragma unroll
            for (int r = 0; r < 8; ++r) {
                int row = mBase + mi * 16 + r + 8 * hf;
                int col = nBlk + nWave + ni * 16 + ln;
                if (col < N) {
                    float val = c[mi][ni].f[r] + (bias ? bias[col] : 0.f);
                    if (EPI == 0) {
                        Cf[(size_t)row * ldc + col] = val;
                    } else if (EPI == 1) {
                        Cb[(size_t)row * ldc + col_off + col] = f2bf(val);
                    } else {
                        size_t o = (size_t)row * ldc + col;
                        Cf[o] = resid[o] + val * ls[col];
                    }
                }
            }
}

// ---------------------------------------------------------------------------
// q/k head-LayerNorm + split + V transpose.  One wave per (token, head).
// q,k -> [bh][n][64] bf16 ;  v -> transposed [bh][64][n] bf16
// ---------------------------------------------------------------------------
__global__ __launch_bounds__(128)
void qkv_ln_kernel(const float* __restrict__ f1,
                   const float* __restrict__ qg, const float* __restrict__ qb,
                   const float* __restrict__ kg, const float* __restrict__ kb,
                   unsigned short* __restrict__ Qo, unsigned short* __restrict__ Ko,
                   unsigned short* __restrict__ VTo) {
    const int gid  = blockIdx.x * 4 + (threadIdx.x >> 5);
    const int lane = threadIdx.x & 31;
    const int t = gid >> 4;           // token
    const int h = gid & 15;
    const int b = t >> 10, n = t & 1023;
    const int bh = b * HEADS + h;
    const float* base = f1 + (size_t)t * NQKV + h * HD;
    const int d0 = lane, d1 = lane + 32;

    // q layernorm
    {
        float q0 = base[d0], q1 = base[d1];
        float s = q0 + q1;
        #pragma unroll
        for (int off = 1; off < 32; off <<= 1) s += __shfl_xor(s, off, 32);
        float mean = s * (1.f / HD);
        float e0 = q0 - mean, e1 = q1 - mean;
        float vs = e0 * e0 + e1 * e1;
        #pragma unroll
        for (int off = 1; off < 32; off <<= 1) vs += __shfl_xor(vs, off, 32);
        float inv = rsqrtf(vs * (1.f / HD) + 1e-5f);
        size_t o = (size_t)(bh * 1024 + n) * HD;
        Qo[o + d0] = f2bf(e0 * inv * qg[d0] + qb[d0]);
        Qo[o + d1] = f2bf(e1 * inv * qg[d1] + qb[d1]);
    }
    // k layernorm
    {
        float q0 = base[DIM + d0], q1 = base[DIM + d1];
        float s = q0 + q1;
        #pragma unroll
        for (int off = 1; off < 32; off <<= 1) s += __shfl_xor(s, off, 32);
        float mean = s * (1.f / HD);
        float e0 = q0 - mean, e1 = q1 - mean;
        float vs = e0 * e0 + e1 * e1;
        #pragma unroll
        for (int off = 1; off < 32; off <<= 1) vs += __shfl_xor(vs, off, 32);
        float inv = rsqrtf(vs * (1.f / HD) + 1e-5f);
        size_t o = (size_t)(bh * 1024 + n) * HD;
        Ko[o + d0] = f2bf(e0 * inv * kg[d0] + kb[d0]);
        Ko[o + d1] = f2bf(e1 * inv * kg[d1] + kb[d1]);
    }
    // v (transposed [d][n] for packed B-fragment loads in attention)
    VTo[(size_t)(bh * HD + d0) * 1024 + n] = f2bf(base[2 * DIM + d0]);
    VTo[(size_t)(bh * HD + d1) * 1024 + n] = f2bf(base[2 * DIM + d1]);
}

// ---------------------------------------------------------------------------
// Exact GELU of mlp_hidden -> bf16, padded K (HIDP)
// ---------------------------------------------------------------------------
__global__ __launch_bounds__(256)
void gelu_kernel(const float* __restrict__ f1, unsigned short* __restrict__ G) {
    size_t i = (size_t)blockIdx.x * 256 + threadIdx.x;
    if (i >= (size_t)NTOK * HIDP) return;
    int t = (int)(i / HIDP);
    int j = (int)(i % HIDP);
    float v = (j < HID) ? f1[(size_t)t * NQKV + 3 * DIM + j] : 0.f;
    float g = 0.5f * v * (1.f + erff(v * 0.70710678118654752f));
    G[i] = f2bf(g);
}

// ---------------------------------------------------------------------------
// Flash attention, one wave = 16 query rows of one (b,h).  Online softmax,
// QK^T and PV both via WMMA bf16; P re-shaped C->A layout through LDS.
// Output written directly as reassembled [B,N,DIM] bf16 for the proj GEMM.
// ---------------------------------------------------------------------------
__global__ __launch_bounds__(128)
void attn_kernel(const unsigned short* __restrict__ Q,
                 const unsigned short* __restrict__ Kb,
                 const unsigned short* __restrict__ Vt,
                 unsigned short* __restrict__ XO) {
    __shared__ unsigned short sP[4][16 * 32];   // per-wave 16x32 P tile
    const int lane = threadIdx.x & 31;
    const int wave = threadIdx.x >> 5;
    const int hf   = lane >> 4;
    const int ln   = lane & 15;
    const int bh   = blockIdx.x;                 // 0..63
    const int m0   = blockIdx.y * 64 + wave * 16;
    const int b = bh >> 4, h = bh & 15;

    // Q fragments (16 rows x 64 K, two 16x32 A-frags)
    AFrag aq[2];
    #pragma unroll
    for (int kt = 0; kt < 2; ++kt) {
        const unsigned short* qp = Q + (size_t)(bh * 1024 + m0 + ln) * HD + kt * 32;
        #pragma unroll
        for (int v = 0; v < 8; ++v) {
            int k = 16 * hf + 2 * v;
            aq[kt].u[v] = *(const u32a*)(qp + k);
        }
    }

    CFrag o[4];
    float mrow[8], lrow[8];
    #pragma unroll
    for (int dt = 0; dt < 4; ++dt)
        #pragma unroll
        for (int r = 0; r < 8; ++r) o[dt].f[r] = 0.f;
    #pragma unroll
    for (int r = 0; r < 8; ++r) { mrow[r] = -1e30f; lrow[r] = 0.f; }

    for (int j = 0; j < 1024 / 32; ++j) {        // key tiles of 32
        CFrag s[2];
        #pragma unroll
        for (int nt = 0; nt < 2; ++nt) {
            AFrag bk0, bk1;
            const unsigned short* kp =
                Kb + (size_t)(bh * 1024 + j * 32 + nt * 16 + ln) * HD + 16 * hf;
            #pragma unroll
            for (int v = 0; v < 8; ++v) {
                bk0.u[v] = *(const u32a*)(kp + 2 * v);
                bk1.u[v] = *(const u32a*)(kp + 32 + 2 * v);
            }
            CFrag t;
            #pragma unroll
            for (int r = 0; r < 8; ++r) t.f[r] = 0.f;
            t.v = __builtin_amdgcn_wmma_f32_16x16x32_bf16(false, aq[0].v, false, bk0.v,
                                                          (short)0, t.v, false, false);
            t.v = __builtin_amdgcn_wmma_f32_16x16x32_bf16(false, aq[1].v, false, bk1.v,
                                                          (short)0, t.v, false, false);
            s[nt] = t;
        }
        // online softmax stats per row (16-lane shuffle reductions stay in-half)
        #pragma unroll
        for (int r = 0; r < 8; ++r) {
            float v0 = s[0].f[r] * 0.125f;       // HEAD_DIM^-0.5
            float v1 = s[1].f[r] * 0.125f;
            float mx = fmaxf(v0, v1);
            #pragma unroll
            for (int off = 1; off < 16; off <<= 1) mx = fmaxf(mx, __shfl_xor(mx, off, 32));
            float nm = fmaxf(mrow[r], mx);
            float al = __expf(mrow[r] - nm);
            mrow[r] = nm;
            float p0 = __expf(v0 - nm);
            float p1 = __expf(v1 - nm);
            int row = r + 8 * hf;
            sP[wave][row * 32 + ln]      = f2bf(p0);
            sP[wave][row * 32 + 16 + ln] = f2bf(p1);
            float ps = p0 + p1;
            #pragma unroll
            for (int off = 1; off < 16; off <<= 1) ps += __shfl_xor(ps, off, 32);
            lrow[r] = lrow[r] * al + ps;
            #pragma unroll
            for (int dt = 0; dt < 4; ++dt) o[dt].f[r] *= al;
        }
        // reload P in A-fragment layout (per-wave LDS region, DS ops in-order)
        AFrag ap;
        #pragma unroll
        for (int v = 0; v < 8; ++v) {
            int k = 16 * hf + 2 * v;
            ap.u[v] = *(const u32a*)(&sP[wave][ln * 32 + k]);
        }
        // O += P(16x32) @ V(32x64)
        #pragma unroll
        for (int dt = 0; dt < 4; ++dt) {
            AFrag bv;
            const unsigned short* vp =
                Vt + (size_t)(bh * HD + dt * 16 + ln) * 1024 + j * 32 + 16 * hf;
            #pragma unroll
            for (int v = 0; v < 8; ++v) bv.u[v] = *(const u32a*)(vp + 2 * v);
            o[dt].v = __builtin_amdgcn_wmma_f32_16x16x32_bf16(false, ap.v, false, bv.v,
                                                              (short)0, o[dt].v, false, false);
        }
    }

    // normalize and store into reassembled [B,N,DIM] layout
    #pragma unroll
    for (int dt = 0; dt < 4; ++dt)
        #pragma unroll
        for (int r = 0; r < 8; ++r) {
            int row = m0 + r + 8 * hf;
            int col = h * HD + dt * 16 + ln;
            XO[(size_t)(b * 1024 + row) * DIM + col] = f2bf(o[dt].f[r] / lrow[r]);
        }
}

// ---------------------------------------------------------------------------
// Host orchestration
// ---------------------------------------------------------------------------
static inline size_t alignup(size_t v) { return (v + 255) & ~(size_t)255; }

extern "C" void kernel_launch(void* const* d_in, const int* in_sizes, int n_in,
                              void* d_out, int out_size, void* d_ws, size_t ws_size,
                              hipStream_t stream) {
    const float* x      = (const float*)d_in[0];
    const float* norm_g = (const float*)d_in[1];
    const float* norm_b = (const float*)d_in[2];
    const float* W1     = (const float*)d_in[3];
    const float* b1     = (const float*)d_in[4];
    const float* qn_g   = (const float*)d_in[5];
    const float* qn_b   = (const float*)d_in[6];
    const float* kn_g   = (const float*)d_in[7];
    const float* kn_b   = (const float*)d_in[8];
    const float* projW  = (const float*)d_in[9];
    const float* projb  = (const float*)d_in[10];
    const float* mlpW   = (const float*)d_in[11];
    const float* W2     = (const float*)d_in[12];
    const float* b2     = (const float*)d_in[13];
    const float* ls_g   = (const float*)d_in[14];
    float* out = (float*)d_out;

    char* ws = (char*)d_ws;
    auto take = [&](size_t bytes) -> char* { char* p = ws; ws += alignup(bytes); return p; };

    unsigned short* w1T   = (unsigned short*)take((size_t)NQKV * DIM * 2);   // [3788][1024]
    unsigned short* projT = (unsigned short*)take((size_t)DIM * DIM * 2);    // [1024][1024]
    unsigned short* mlpT  = (unsigned short*)take((size_t)DIM * HIDP * 2);   // [1024][736]
    unsigned short* w2T   = (unsigned short*)take((size_t)DIM * D2 * 2);     // [1024][2048]
    unsigned short* nxbf  = (unsigned short*)take((size_t)NTOK * DIM * 2);   // LN(x); reused as XO
    float*          f1    = (float*)take((size_t)NTOK * NQKV * 4);
    unsigned short* qbf   = (unsigned short*)take((size_t)64 * 1024 * HD * 2);
    unsigned short* kbf   = (unsigned short*)take((size_t)64 * 1024 * HD * 2);
    unsigned short* vT    = (unsigned short*)take((size_t)64 * HD * 1024 * 2);
    unsigned short* gbf   = (unsigned short*)take((size_t)NTOK * HIDP * 2);
    unsigned short* cat   = (unsigned short*)take((size_t)NTOK * D2 * 2);

    // 1) weights -> bf16, transposed [N][Kp]
    cvt_transpose_kernel<<<(unsigned)(((size_t)NQKV * DIM + 255) / 256), 256, 0, stream>>>(
        W1, w1T, DIM, NQKV, DIM);
    cvt_transpose_kernel<<<(unsigned)(((size_t)DIM * DIM + 255) / 256), 256, 0, stream>>>(
        projW, projT, DIM, DIM, DIM);
    cvt_transpose_kernel<<<(unsigned)(((size_t)DIM * HIDP + 255) / 256), 256, 0, stream>>>(
        mlpW, mlpT, HID, DIM, HIDP);
    cvt_transpose_kernel<<<(unsigned)(((size_t)DIM * D2 + 255) / 256), 256, 0, stream>>>(
        W2, w2T, D2, DIM, D2);

    // 2) LayerNorm(x) -> bf16
    ln_kernel<<<NTOK / 8, 256, 0, stream>>>(x, norm_g, norm_b, nxbf);

    // 3) f1 = LN(x) @ W1 + b1   (fp32)
    gemm_bf16_kernel<0><<<dim3((NQKV + 127) / 128, NTOK / 128), 256, 0, stream>>>(
        nxbf, DIM, w1T, DIM, b1, NQKV, DIM, f1, nullptr, NQKV, 0, nullptr, nullptr);

    // 4) q/k head-LN, split, V transpose
    qkv_ln_kernel<<<(NTOK * HEADS) / 4, 128, 0, stream>>>(
        f1, qn_g, qn_b, kn_g, kn_b, qbf, kbf, vT);

    // 5) GELU(mlp_hidden) -> bf16 (K padded to 736)
    gelu_kernel<<<(unsigned)(((size_t)NTOK * HIDP + 255) / 256), 256, 0, stream>>>(f1, gbf);

    // 6) attention (XO reuses nxbf buffer — LN output is dead after step 3)
    unsigned short* xo = nxbf;
    attn_kernel<<<dim3(64, 1024 / 64), 128, 0, stream>>>(qbf, kbf, vT, xo);

    // 7) attn proj -> concat[:, :1024] (bf16, +projb)
    gemm_bf16_kernel<1><<<dim3(DIM / 128, NTOK / 128), 256, 0, stream>>>(
        xo, DIM, projT, DIM, projb, DIM, DIM, nullptr, cat, D2, 0, nullptr, nullptr);

    // 8) mlp out -> concat[:, 1024:] (bf16, no bias)
    gemm_bf16_kernel<1><<<dim3(DIM / 128, NTOK / 128), 256, 0, stream>>>(
        gbf, HIDP, mlpT, HIDP, nullptr, DIM, HIDP, nullptr, cat, D2, DIM, nullptr, nullptr);

    // 9) out = x + (concat @ W2 + b2) * ls_g
    gemm_bf16_kernel<2><<<dim3(DIM / 128, NTOK / 128), 256, 0, stream>>>(
        cat, D2, w2T, D2, b2, DIM, D2, out, nullptr, DIM, 0, x, ls_g);

    (void)in_sizes; (void)n_in; (void)out_size; (void)ws_size;
}
